// GQAAttention_1924145348706
// MI455X (gfx1250) — compile-verified, matrix operand order checked by reading, not run on hardware
//
#include <hip/hip_runtime.h>
#include <hip/hip_bf16.h>

#define B_   2
#define T_   2048
#define DM   1024
#define NH   16
#define NKV  4
#define DH   64
#define GRP  4

typedef __attribute__((ext_vector_type(16))) _Float16    v16h;
typedef __attribute__((ext_vector_type(8)))  float       v8f;
typedef __attribute__((ext_vector_type(4)))  unsigned int u32x4;

union HFrag { v16h v; _Float16 e[16]; u32x4 q[2]; };

// Load a 32-element (k-chunk) f16 A/B fragment. Lane layout per CDNA5 ISA
// 16-bit 16x32 table: half h covers dwords [4h..4h+3] and [8+4h..8+4h+3].
__device__ __forceinline__ v16h load_frag_f16(const _Float16* rowChunk, int half) {
  const u32x4* p = (const u32x4*)rowChunk;
  HFrag f;
  f.q[0] = p[half];
  f.q[1] = p[2 + half];
  return f.v;
}

__device__ __forceinline__ v16h load_frag_f32cvt(const float* rowChunk, int half) {
  HFrag f;
  const int k0 = 8 * half, k1 = 16 + 8 * half;
#pragma unroll
  for (int j = 0; j < 8; ++j) f.e[j]     = (_Float16)rowChunk[k0 + j];
#pragma unroll
  for (int j = 0; j < 8; ++j) f.e[8 + j] = (_Float16)rowChunk[k1 + j];
  return f.v;
}

__device__ __forceinline__ v8f wmma16(v16h a, v16h b, v8f c) {
  return __builtin_amdgcn_wmma_f32_16x16x32_f16(false, a, false, b, (short)0, c, false, false);
}

__device__ __forceinline__ float half16_max(float v) {
  v = fmaxf(v, __shfl_xor(v, 1, 32));
  v = fmaxf(v, __shfl_xor(v, 2, 32));
  v = fmaxf(v, __shfl_xor(v, 4, 32));
  v = fmaxf(v, __shfl_xor(v, 8, 32));
  return v;
}
__device__ __forceinline__ float half16_sum(float v) {
  v += __shfl_xor(v, 1, 32);
  v += __shfl_xor(v, 2, 32);
  v += __shfl_xor(v, 4, 32);
  v += __shfl_xor(v, 8, 32);
  return v;
}

// ---------------- Kernel 0: weight transpose + f32->f16 ----------------
__global__ void k_transpose_cvt(const float* __restrict__ src,
                                _Float16* __restrict__ dst, int K, int N) {
  int idx = blockIdx.x * blockDim.x + threadIdx.x;
  if (idx >= K * N) return;
  int k = idx / N, n = idx % N;
  dst[(size_t)n * K + k] = (_Float16)src[idx];
}

// ---------------- Kernel 1: fused QKV projection + RoPE ----------------
// One wave = 16 tokens x one head's full 64 dims (4 accum tiles).
__global__ void k_qkv_rope(const float* __restrict__ x,
                           const _Float16* __restrict__ wqT,   // [1024][1024]
                           const _Float16* __restrict__ wkT,   // [256][1024]
                           const _Float16* __restrict__ wvT,   // [256][1024]
                           _Float16* __restrict__ Q,           // [B][NH][T][DH]
                           _Float16* __restrict__ K,           // [B][NKV][T][DH]
                           _Float16* __restrict__ Vt)          // [B][NKV][DH][T]
{
  const int wave = threadIdx.x >> 5;
  const int lane = threadIdx.x & 31;
  const int half = lane >> 4;
  const int n    = lane & 15;
  const int mB   = 8 * half;

  const int job     = blockIdx.x * 8 + wave;   // 256 rowTiles * 24 head-jobs
  const int rowTile = job / 24;
  const int sub     = job % 24;

  const int r0 = rowTile * 16;   // base over B*T flattened rows
  const int b  = r0 / T_;
  const int t0 = r0 % T_;

  const _Float16* wT;
  int hh, type;
  if (sub < 16)      { type = 0; hh = sub;      wT = wqT; }
  else if (sub < 20) { type = 1; hh = sub - 16; wT = wkT; }
  else               { type = 2; hh = sub - 20; wT = wvT; }

  const float* xrow = x + (size_t)(r0 + n) * DM;   // A row = lane%16

  v8f acc[4] = {};
  for (int c = 0; c < DM / 32; ++c) {
    v16h a = load_frag_f32cvt(xrow + 32 * c, half);
#pragma unroll
    for (int tp = 0; tp < 4; ++tp) {
      const _Float16* wrow = wT + (size_t)(hh * 64 + 16 * tp + n) * DM + 32 * c;
      acc[tp] = wmma16(a, load_frag_f16(wrow, half), acc[tp]);
    }
  }

  if (type == 2) {                                   // V: store transposed [DH][T]
    _Float16* base = Vt + (size_t)(b * NKV + hh) * DH * T_;
#pragma unroll
    for (int tp = 0; tp < 4; ++tp) {
      int d = 16 * tp + n;
#pragma unroll
      for (int r = 0; r < 8; ++r)
        base[(size_t)d * T_ + (t0 + mB + r)] = (_Float16)acc[tp][r];
    }
  } else {                                           // Q/K: RoPE then store rows
    _Float16* base = (type == 0)
      ? Q + (size_t)(b * NH  + hh) * T_ * DH
      : K + (size_t)(b * NKV + hh) * T_ * DH;
#pragma unroll
    for (int tp = 0; tp < 2; ++tp) {
      int d = 16 * tp + n;                           // 0..31 (pair index)
      float invf = __expf(-(float)d * (9.210340371976184f / 32.0f)); // 10000^(-d/32)
#pragma unroll
      for (int r = 0; r < 8; ++r) {
        int t = t0 + mB + r;
        float ang = (float)t * invf;
        float cs = __cosf(ang), sn = __sinf(ang);
        float x1 = acc[tp][r], x2 = acc[tp + 2][r];
        _Float16* rowp = base + (size_t)t * DH;
        rowp[d]      = (_Float16)(x1 * cs - x2 * sn);
        rowp[d + 32] = (_Float16)(x2 * cs + x1 * sn);
      }
    }
  }
}

// ---------------- Kernel 2: causal flash attention ----------------
// One wave = 16 query rows; 8 waves/block; block covers 128 queries of one (b,h).
__global__ void k_attn(const _Float16* __restrict__ Q,
                       const _Float16* __restrict__ Kc,   // [B][NKV][T][DH]
                       const _Float16* __restrict__ Vt,   // [B][NKV][DH][T]
                       _Float16* __restrict__ Ocat)       // [B*T][DM] head-concat
{
  __shared__ __align__(16) _Float16 Plds[8][16][40];      // per-wave P staging
  const int wave = threadIdx.x >> 5;
  const int lane = threadIdx.x & 31;
  const int half = lane >> 4;
  const int n    = lane & 15;
  const int mB   = 8 * half;

  const int qg = blockIdx.x & 15;
  const int bh = blockIdx.x >> 4;
  const int b = bh >> 4, h = bh & 15;
  const int kvh = h / GRP;
  const int q0 = qg * 128 + wave * 16;

  const _Float16* Qrow = Q + ((size_t)(b * NH + h) * T_ + q0 + n) * DH;
  v16h aq0 = load_frag_f16(Qrow, half);        // k = 0..31 of head dim
  v16h aq1 = load_frag_f16(Qrow + 32, half);   // k = 32..63

  const _Float16* Kbase = Kc + (size_t)(b * NKV + kvh) * T_ * DH;
  const _Float16* Vbase = Vt + (size_t)(b * NKV + kvh) * DH * T_;

  v8f o0 = {}, o1 = {}, o2 = {}, o3 = {};
  float m_i[8], l_i[8];
#pragma unroll
  for (int r = 0; r < 8; ++r) { m_i[r] = -3.0e38f; l_i[r] = 0.0f; }

  const int limit = q0 + 16;
  const int nch = (limit + 31) >> 5;

  for (int ci = 0; ci < nch; ++ci) {
    const int kc = ci << 5;
    const _Float16* Kr0 = Kbase + (size_t)(kc + n) * DH;
    const _Float16* Kr1 = Kbase + (size_t)(kc + 16 + n) * DH;

    v8f s0 = {}, s1 = {};
    s0 = wmma16(aq0, load_frag_f16(Kr0,      half), s0);
    s0 = wmma16(aq1, load_frag_f16(Kr0 + 32, half), s0);
    s1 = wmma16(aq0, load_frag_f16(Kr1,      half), s1);
    s1 = wmma16(aq1, load_frag_f16(Kr1 + 32, half), s1);

#pragma unroll
    for (int r = 0; r < 8; ++r) {
      const int qi = q0 + mB + r;
      float v0 = (kc + n      <= qi) ? s0[r] * 0.125f : -3.0e38f;
      float v1 = (kc + 16 + n <= qi) ? s1[r] * 0.125f : -3.0e38f;
      float rm   = half16_max(fmaxf(v0, v1));
      float mnew = fmaxf(m_i[r], rm);
      float p0 = __expf(v0 - mnew);
      float p1 = __expf(v1 - mnew);
      float alpha = __expf(m_i[r] - mnew);
      float ps = half16_sum(p0 + p1);
      l_i[r] = l_i[r] * alpha + ps;
      m_i[r] = mnew;
      o0[r] *= alpha; o1[r] *= alpha; o2[r] *= alpha; o3[r] *= alpha;
      Plds[wave][mB + r][n]      = (_Float16)p0;   // C-layout -> LDS
      Plds[wave][mB + r][n + 16] = (_Float16)p1;
    }
    // wave-local LDS write->read fence (waves are divergent; no block barrier)
    asm volatile("s_wait_dscnt 0x0" ::: "memory");

    v16h ap = load_frag_f16(&Plds[wave][n][0], half);  // A-layout P
    o0 = wmma16(ap, load_frag_f16(Vbase + (size_t)(n)      * T_ + kc, half), o0);
    o1 = wmma16(ap, load_frag_f16(Vbase + (size_t)(16 + n) * T_ + kc, half), o1);
    o2 = wmma16(ap, load_frag_f16(Vbase + (size_t)(32 + n) * T_ + kc, half), o2);
    o3 = wmma16(ap, load_frag_f16(Vbase + (size_t)(48 + n) * T_ + kc, half), o3);
  }

#pragma unroll
  for (int r = 0; r < 8; ++r) {
    float inv = 1.0f / l_i[r];
    size_t row = (size_t)(b * T_ + q0 + mB + r) * DM + h * DH;
    Ocat[row +  0 + n] = (_Float16)(o0[r] * inv);
    Ocat[row + 16 + n] = (_Float16)(o1[r] * inv);
    Ocat[row + 32 + n] = (_Float16)(o2[r] * inv);
    Ocat[row + 48 + n] = (_Float16)(o3[r] * inv);
  }
}

// ---------------- Kernel 3: output projection ----------------
__global__ void k_oproj(const _Float16* __restrict__ Ocat,  // [B*T][DM]
                        const _Float16* __restrict__ woT,   // [DM][DM] = [n][k]
                        float* __restrict__ out)            // [B*T][DM]
{
  const int wave = threadIdx.x >> 5;
  const int lane = threadIdx.x & 31;
  const int half = lane >> 4;
  const int n    = lane & 15;
  const int mB   = 8 * half;

  const int job = blockIdx.x * 8 + wave;  // 256 rowTiles * 64 colTiles
  const int rowTile = job >> 6;
  const int ct = job & 63;
  const int r0 = rowTile * 16;

  const _Float16* arow = Ocat + (size_t)(r0 + n) * DM;
  const _Float16* brow = woT  + (size_t)(ct * 16 + n) * DM;

  v8f acc = {};
  for (int c = 0; c < DM / 32; ++c)
    acc = wmma16(load_frag_f16(arow + 32 * c, half),
                 load_frag_f16(brow + 32 * c, half), acc);

#pragma unroll
  for (int r = 0; r < 8; ++r)
    out[(size_t)(r0 + mB + r) * DM + ct * 16 + n] = acc[r];
}

// ---------------- Host-side launch ----------------
extern "C" void kernel_launch(void* const* d_in, const int* in_sizes, int n_in,
                              void* d_out, int out_size, void* d_ws, size_t ws_size,
                              hipStream_t stream) {
  const float* x  = (const float*)d_in[0];
  const float* wq = (const float*)d_in[1];
  const float* wk = (const float*)d_in[2];
  const float* wv = (const float*)d_in[3];
  const float* wo = (const float*)d_in[4];
  float* out = (float*)d_out;

  _Float16* ws = (_Float16*)d_ws;
  size_t off = 0;
  _Float16* Q    = ws + off; off += (size_t)B_ * NH  * T_ * DH;   // 4,194,304
  _Float16* K    = ws + off; off += (size_t)B_ * NKV * T_ * DH;   // 1,048,576
  _Float16* Vt   = ws + off; off += (size_t)B_ * NKV * DH * T_;   // 1,048,576
  _Float16* Ocat = ws + off; off += (size_t)B_ * T_ * DM;         // 4,194,304
  _Float16* wqT  = ws + off; off += (size_t)DM * DM;              // 1,048,576
  _Float16* wkT  = ws + off; off += (size_t)DM * (NKV * DH);      //   262,144
  _Float16* wvT  = ws + off; off += (size_t)DM * (NKV * DH);      //   262,144
  _Float16* woT  = ws + off; off += (size_t)DM * DM;              // 1,048,576
  // total ~26.2 MB of d_ws

  k_transpose_cvt<<<(DM * DM + 255) / 256, 256, 0, stream>>>(wq, wqT, DM, DM);
  k_transpose_cvt<<<(DM * NKV * DH + 255) / 256, 256, 0, stream>>>(wk, wkT, DM, NKV * DH);
  k_transpose_cvt<<<(DM * NKV * DH + 255) / 256, 256, 0, stream>>>(wv, wvT, DM, NKV * DH);
  k_transpose_cvt<<<(DM * DM + 255) / 256, 256, 0, stream>>>(wo, woT, DM, DM);

  // 256 row-tiles * 24 head-jobs = 6144 waves / 8 per block
  k_qkv_rope<<<768, 256, 0, stream>>>(x, wqT, wkT, wvT, Q, K, Vt);

  // B*NH * (T/128) = 512 blocks, 8 waves each (one 16-query tile per wave)
  k_attn<<<512, 256, 0, stream>>>(Q, K, Vt, Ocat);

  // 256 row-tiles * 64 col-tiles = 16384 waves / 8 per block
  k_oproj<<<2048, 256, 0, stream>>>(Ocat, woT, out);
}
